// DeepSeekMoE_47158740910261
// MI455X (gfx1250) — matrix operand order repaired and checked by reading, not compiled, not run on hardware
//
#include <hip/hip_runtime.h>
#include <hip/hip_bf16.h>

// ---------------------------------------------------------------------------
// DeepSeek-style MoE block for MI455X (gfx1250, wave32, WMMA, async-LDS).
//
//   out = sq_relu(x@Ws1)@Ws2 + sum_e trunc(softmax(top2(x@Wr)))[e]*(sq_relu(x@W1_e)@W2_e)
//
// trunc() of a 2-way softmax is identically 0 for this input distribution, so
// the routed term is a guarded never-taken fallback; the hot path is the two
// shared-expert GEMMs run as bf16 WMMA (f32 accumulate).  All f32->bf16
// conversion + B transposition is hoisted into cheap bandwidth-bound prep
// kernels (~18 MB traffic ~ 1 us at 23.3 TB/s) so the GEMM K-loop is pure
// async-global->LDS copies + ds_load_b128 + v_wmma.
// ---------------------------------------------------------------------------

#define D_MODEL    1024
#define EXPERT_DIM 512
#define N_EXPERTS  8
#define T_TOKENS   4096

typedef __attribute__((ext_vector_type(16))) __bf16 v16bf;
typedef __attribute__((ext_vector_type(8)))  __bf16 v8bf;
typedef __attribute__((ext_vector_type(8)))  float  v8f;
typedef __attribute__((ext_vector_type(4)))  int    v4i;

#define HAVE_ASYNC_LDS __has_builtin(__builtin_amdgcn_global_load_async_to_lds_b128)

// async builtin wants int4* in explicit address spaces
#define GPTR_V4I(p) ((__attribute__((address_space(1))) v4i*)(p))
#define LPTR_V4I(p) ((__attribute__((address_space(3))) v4i*)(p))

__device__ inline unsigned short f2bf(float f) {
    union { float f; unsigned u; } v; v.f = f;
    unsigned r = v.u + 0x7FFFu + ((v.u >> 16) & 1u);   // round-to-nearest-even
    return (unsigned short)(r >> 16);
}

// ---------------------------------------------------------------------------
// Prep kernels (bandwidth-bound, negligible vs GEMMs)
// ---------------------------------------------------------------------------
__global__ __launch_bounds__(256)
void cvt_bf16_kernel(const float* __restrict__ src, unsigned short* __restrict__ dst,
                     int n4)   // n4 = count/4
{
    int i = blockIdx.x * blockDim.x + threadIdx.x;
    if (i >= n4) return;
    float4 v = ((const float4*)src)[i];
    uint2 o;
    o.x = (unsigned)f2bf(v.x) | ((unsigned)f2bf(v.y) << 16);
    o.y = (unsigned)f2bf(v.z) | ((unsigned)f2bf(v.w) << 16);
    ((uint2*)dst)[i] = o;
}

// dst[n][k] = bf16(src[k][n])   (src: [K][N] f32, dst: [N][K] bf16)
__global__ __launch_bounds__(256)
void transpose_cvt_kernel(const float* __restrict__ src, unsigned short* __restrict__ dst,
                          int K, int N)
{
    int i = blockIdx.x * blockDim.x + threadIdx.x;
    if (i >= K * N) return;
    int k = i / N, n = i - k * N;          // thread-fast dim = n -> coalesced reads
    dst[(size_t)n * K + k] = f2bf(src[i]);
}

// ---------------------------------------------------------------------------
// Tiled bf16 GEMM: C[M,N] = epilogue(A[M,K] @ Bt[N,K]^T)
//   A, Bt: bf16 row-major (Bt is B pre-transposed -> both tiles stage as
//   plain row chunks, enabling GLOBAL_LOAD_ASYNC_TO_LDS_B128).
//   epilogue: SQRELU_OUT=1 -> relu(x)^2, store bf16 ; 0 -> store f32
// Block: 256 threads (8 waves, 2x4 wave grid), tile 128x128, K-step 32.
// Each wave: 4x2 grid of v_wmma_f32_16x16x32_bf16 accumulators.
// ---------------------------------------------------------------------------
template<bool SQRELU_OUT>
__global__ __launch_bounds__(256)
void wmma_gemm_bf16_kernel(const unsigned short* __restrict__ A,
                           const unsigned short* __restrict__ Bt,
                           void* __restrict__ Cp, int M, int N, int K)
{
    __shared__ alignas(16) unsigned short sA [2][128][32];   // [m][k] bf16
    __shared__ alignas(16) unsigned short sBt[2][128][32];   // [n][k] bf16

    const int tid   = threadIdx.x;
    const int lane  = tid & 31;
    const int wave  = tid >> 5;
    const int waveM = wave >> 2;        // 0..1
    const int waveN = wave & 3;         // 0..3
    const int half  = lane >> 4;        // 0..1
    const int l16   = lane & 15;        // 0..15
    const int mBase = blockIdx.y * 128;
    const int nBase = blockIdx.x * 128;

    v8f acc[4][2];
#pragma unroll
    for (int i = 0; i < 4; ++i)
#pragma unroll
        for (int j = 0; j < 2; ++j)
#pragma unroll
            for (int r = 0; r < 8; ++r) acc[i][j][r] = 0.0f;

    // tile staging: 128 rows x 32 bf16 = 512 x 16B chunks; 2 chunks/thread/tile
    const int cr0 = tid >> 2,         cq0 = (tid & 3) * 8;          // chunk 0
    const int cr1 = (tid + 256) >> 2, cq1 = ((tid + 256) & 3) * 8;  // chunk 1

#if HAVE_ASYNC_LDS
    auto issueTiles = [&](int kB, int buf) {
        __builtin_amdgcn_global_load_async_to_lds_b128(
            GPTR_V4I(A  + (size_t)(mBase + cr0) * K + kB + cq0),
            LPTR_V4I(&sA [buf][cr0][cq0]), 0, 0);
        __builtin_amdgcn_global_load_async_to_lds_b128(
            GPTR_V4I(A  + (size_t)(mBase + cr1) * K + kB + cq1),
            LPTR_V4I(&sA [buf][cr1][cq1]), 0, 0);
        __builtin_amdgcn_global_load_async_to_lds_b128(
            GPTR_V4I(Bt + (size_t)(nBase + cr0) * K + kB + cq0),
            LPTR_V4I(&sBt[buf][cr0][cq0]), 0, 0);
        __builtin_amdgcn_global_load_async_to_lds_b128(
            GPTR_V4I(Bt + (size_t)(nBase + cr1) * K + kB + cq1),
            LPTR_V4I(&sBt[buf][cr1][cq1]), 0, 0);
    };
    auto waitAsync = [&]() {
#if __has_builtin(__builtin_amdgcn_s_wait_asynccnt)
        __builtin_amdgcn_s_wait_asynccnt(0);
#else
        asm volatile("s_wait_asynccnt 0x0" ::: "memory");
#endif
    };
#else
    uint4 rA[2], rB[2];
    auto loadTiles = [&](int kB) {
        rA[0] = *(const uint4*)(A  + (size_t)(mBase + cr0) * K + kB + cq0);
        rA[1] = *(const uint4*)(A  + (size_t)(mBase + cr1) * K + kB + cq1);
        rB[0] = *(const uint4*)(Bt + (size_t)(nBase + cr0) * K + kB + cq0);
        rB[1] = *(const uint4*)(Bt + (size_t)(nBase + cr1) * K + kB + cq1);
    };
    auto storeTiles = [&](int buf) {
        *(uint4*)&sA [buf][cr0][cq0] = rA[0];
        *(uint4*)&sA [buf][cr1][cq1] = rA[1];
        *(uint4*)&sBt[buf][cr0][cq0] = rB[0];
        *(uint4*)&sBt[buf][cr1][cq1] = rB[1];
    };
#endif

    const int KT = K >> 5;

#if HAVE_ASYNC_LDS
    issueTiles(0, 0);
    waitAsync();
    __syncthreads();
#else
    loadTiles(0); storeTiles(0);
    __syncthreads();
#endif

    for (int kt = 0; kt < KT; ++kt) {
        const int cur = kt & 1;
#if HAVE_ASYNC_LDS
        if (kt + 1 < KT) issueTiles((kt + 1) << 5, (kt + 1) & 1);
#else
        if (kt + 1 < KT) loadTiles((kt + 1) << 5);
#endif
        if (kt + 2 < KT) {                        // L2 prefetch (global_prefetch_b8)
            __builtin_prefetch(A  + (size_t)(mBase + cr0) * K + ((kt + 2) << 5) + cq0, 0, 0);
            __builtin_prefetch(Bt + (size_t)(nBase + cr0) * K + ((kt + 2) << 5) + cq0, 0, 0);
        }

        // A fragment (16x32 bf16): vec elems 0..7 <-> K=8*half+0..7,
        // elems 8..15 <-> K=16+8*half+0..7 (row = l16); both contiguous b128.
        v16bf afr[4];
#pragma unroll
        for (int i = 0; i < 4; ++i) {
            const unsigned short* rp = &sA[cur][waveM * 64 + i * 16 + l16][0];
            v8bf lo = *(const v8bf*)(rp + 8 * half);
            v8bf hi = *(const v8bf*)(rp + 16 + 8 * half);
#pragma unroll
            for (int e = 0; e < 8; ++e) { afr[i][e] = lo[e]; afr[i][8 + e] = hi[e]; }
        }
        // B fragment (32x16 bf16): elems 0..7 <-> K=8*half+0..7, elems 8..15 <-> K=16+8*half+0..7 (col = l16).
        v16bf bfr[2];
#pragma unroll
        for (int j = 0; j < 2; ++j) {
            const unsigned short* rp = &sBt[cur][waveN * 32 + j * 16 + l16][0];
            v8bf lo = *(const v8bf*)(rp + 8 * half);
            v8bf hi = *(const v8bf*)(rp + 16 + 8 * half);
#pragma unroll
            for (int e = 0; e < 8; ++e) { bfr[j][e] = lo[e]; bfr[j][8 + e] = hi[e]; }
        }

#pragma unroll
        for (int i = 0; i < 4; ++i)
#pragma unroll
            for (int j = 0; j < 2; ++j)
                acc[i][j] = __builtin_amdgcn_wmma_f32_16x16x32_bf16(
                    false, afr[i], false, bfr[j], (short)0, acc[i][j], false, false);

#if HAVE_ASYNC_LDS
        waitAsync();          // next tile fully in LDS
#else
        if (kt + 1 < KT) storeTiles((kt + 1) & 1);
#endif
        __syncthreads();
    }

    // Epilogue. D layout: lane L, vgpr r -> M = r + 8*half, N = l16.
#pragma unroll
    for (int i = 0; i < 4; ++i)
#pragma unroll
        for (int j = 0; j < 2; ++j)
#pragma unroll
            for (int r = 0; r < 8; ++r) {
                int gm = mBase + waveM * 64 + i * 16 + r + 8 * half;
                int gn = nBase + waveN * 32 + j * 16 + l16;
                float v = acc[i][j][r];
                if constexpr (SQRELU_OUT) {
                    v = v > 0.0f ? v * v : 0.0f;
                    ((unsigned short*)Cp)[(size_t)gm * N + gn] = f2bf(v);
                } else {
                    ((float*)Cp)[(size_t)gm * N + gn] = v;
                }
            }
}

// ---------------------------------------------------------------------------
// Router: logits = x@Wr, top-2, softmax(fp32), trunc -> prob[T,E]
// ---------------------------------------------------------------------------
__global__ __launch_bounds__(256)
void router_kernel(const float* __restrict__ x, const float* __restrict__ Wr,
                   float* __restrict__ prob)
{
    int t = blockIdx.x * blockDim.x + threadIdx.x;
    if (t >= T_TOKENS) return;
    const float* xr = x + (size_t)t * D_MODEL;
    float acc[N_EXPERTS];
#pragma unroll
    for (int e = 0; e < N_EXPERTS; ++e) acc[e] = 0.0f;
    for (int k = 0; k < D_MODEL; ++k) {
        float xv = xr[k];
#pragma unroll
        for (int e = 0; e < N_EXPERTS; ++e) acc[e] += xv * Wr[k * N_EXPERTS + e];
    }
    int i0 = 0; float v0 = acc[0];
#pragma unroll
    for (int e = 1; e < N_EXPERTS; ++e) if (acc[e] > v0) { v0 = acc[e]; i0 = e; }
    int i1 = -1; float v1 = -3.4e38f;
#pragma unroll
    for (int e = 0; e < N_EXPERTS; ++e)
        if (e != i0 && acc[e] > v1) { v1 = acc[e]; i1 = e; }
    float e0 = 1.0f;                       // exp(v0 - max) == exp(0)
    float e1 = expf(v1 - v0);
    float s  = e0 + e1;
    float w0 = truncf(e0 / s);
    float w1 = truncf(e1 / s);
    float p[N_EXPERTS];
#pragma unroll
    for (int e = 0; e < N_EXPERTS; ++e) p[e] = 0.0f;
    p[i0] += w0; p[i1] += w1;
#pragma unroll
    for (int e = 0; e < N_EXPERTS; ++e) prob[(size_t)t * N_EXPERTS + e] = p[e];
}

// ---------------------------------------------------------------------------
// Routed-expert fallback: out[t] += prob[t,e] * (sq_relu(x[t]@W1_e)@W2_e).
// prob is identically zero here (trunc of a 2-way softmax), so every thread
// early-exits; exists only for input-generality.
// ---------------------------------------------------------------------------
__global__ __launch_bounds__(256)
void routed_fallback_kernel(const float* __restrict__ x, const float* __restrict__ W1,
                            const float* __restrict__ W2, const float* __restrict__ prob,
                            float* __restrict__ out)
{
    int t = blockIdx.x * blockDim.x + threadIdx.x;
    if (t >= T_TOKENS) return;
    float p[N_EXPERTS]; bool any = false;
#pragma unroll
    for (int e = 0; e < N_EXPERTS; ++e) { p[e] = prob[(size_t)t * N_EXPERTS + e]; any |= (p[e] != 0.0f); }
    if (!any) return;

    const float* xr = x + (size_t)t * D_MODEL;
    float h[EXPERT_DIM];
    for (int e = 0; e < N_EXPERTS; ++e) {
        if (p[e] == 0.0f) continue;
        const float* w1 = W1 + (size_t)e * D_MODEL * EXPERT_DIM;
        for (int f = 0; f < EXPERT_DIM; ++f) {
            float a = 0.0f;
            for (int k = 0; k < D_MODEL; ++k) a += xr[k] * w1[(size_t)k * EXPERT_DIM + f];
            h[f] = a > 0.0f ? a * a : 0.0f;
        }
        const float* w2 = W2 + (size_t)e * EXPERT_DIM * D_MODEL;
        for (int dc = 0; dc < D_MODEL; ++dc) {
            float a = 0.0f;
            for (int f = 0; f < EXPERT_DIM; ++f) a += h[f] * w2[(size_t)f * D_MODEL + dc];
            out[(size_t)t * D_MODEL + dc] += p[e] * a;
        }
    }
}

// ---------------------------------------------------------------------------
extern "C" void kernel_launch(void* const* d_in, const int* in_sizes, int n_in,
                              void* d_out, int out_size, void* d_ws, size_t ws_size,
                              hipStream_t stream)
{
    const float* x   = (const float*)d_in[0];   // [4096,1024]
    const float* Ws1 = (const float*)d_in[1];   // [1024,512]
    const float* Ws2 = (const float*)d_in[2];   // [512,1024]
    const float* W1  = (const float*)d_in[3];   // [8,1024,512]
    const float* W2  = (const float*)d_in[4];   // [8,512,1024]
    const float* Wr  = (const float*)d_in[5];   // [1024,8]
    float* out = (float*)d_out;

    // workspace layout (bytes)
    char* ws = (char*)d_ws;
    unsigned short* Xb   = (unsigned short*)(ws);                        //  8 MB bf16 [4096,1024]
    unsigned short* H    = (unsigned short*)(ws + (8u << 20));           //  4 MB bf16 [4096,512]
    unsigned short* Bt1  = (unsigned short*)(ws + (12u << 20));          //  1 MB bf16 [512,1024]  = Ws1^T
    unsigned short* Bt2  = (unsigned short*)(ws + (13u << 20));          //  1 MB bf16 [1024,512]  = Ws2^T
    float*          prob = (float*)         (ws + (14u << 20));          //  128 KB f32 [4096,8]

    // prep: conversions + transposes (bandwidth-bound)
    cvt_bf16_kernel<<<(T_TOKENS * D_MODEL / 4 + 255) / 256, 256, 0, stream>>>(
        x, Xb, T_TOKENS * D_MODEL / 4);
    transpose_cvt_kernel<<<(D_MODEL * EXPERT_DIM + 255) / 256, 256, 0, stream>>>(
        Ws1, Bt1, D_MODEL, EXPERT_DIM);
    transpose_cvt_kernel<<<(EXPERT_DIM * D_MODEL + 255) / 256, 256, 0, stream>>>(
        Ws2, Bt2, EXPERT_DIM, D_MODEL);

    router_kernel<<<T_TOKENS / 256, 256, 0, stream>>>(x, Wr, prob);

    // H = bf16( sq_relu(Xb @ Ws1) )
    dim3 g1(EXPERT_DIM / 128, T_TOKENS / 128);
    wmma_gemm_bf16_kernel<true><<<g1, 256, 0, stream>>>(Xb, Bt1, H,
                                                        T_TOKENS, EXPERT_DIM, D_MODEL);
    // out = H @ Ws2 (f32)
    dim3 g2(D_MODEL / 128, T_TOKENS / 128);
    wmma_gemm_bf16_kernel<false><<<g2, 256, 0, stream>>>(H, Bt2, out,
                                                         T_TOKENS, D_MODEL, EXPERT_DIM);

    routed_fallback_kernel<<<T_TOKENS / 256, 256, 0, stream>>>(x, W1, W2, prob, out);
}
